// LSTM_24034636988825
// MI455X (gfx1250) — matrix operand order repaired and checked by reading, not compiled, not run on hardware
//
#include <hip/hip_runtime.h>

typedef __attribute__((ext_vector_type(16))) __bf16 v16bf;
typedef __attribute__((ext_vector_type(8)))  float  v8f;
typedef __attribute__((ext_vector_type(4)))  unsigned int u32x4;
typedef __attribute__((ext_vector_type(8)))  unsigned int u32x8;

#define K_DIM 512
#define NG 1024          // 4*H
#define BROWS 64         // batch rows per workgroup
#define THREADS 512      // 16 waves
#define LDS_A_STRIDE 40  // halves (80B rows, 16B aligned)
#define LDS_B_STRIDE 40  // halves (TDM pad: 16dw data + 4dw pad)
#define GATE_STRIDE 1032 // floats
#define A_BYTES 5120     // 64 * 40 * 2
#define B_BYTES 81920    // 1024 * 40 * 2
#define B_OFF 12288
#define SMEM_BYTES (BROWS * GATE_STRIDE * 4)   // 264192 (staging 176128 overlays it)
#define BH 16777216      // 65536*256

__device__ __forceinline__ unsigned short f2bf(float f) {
    unsigned int u = __float_as_uint(f);
    u = u + 0x7FFFu + ((u >> 16) & 1u);   // round-to-nearest-even
    return (unsigned short)(u >> 16);
}

__device__ __forceinline__ void ld8(const float* p, float* d) {
    float4 a = *(const float4*)p;
    float4 b = *(const float4*)(p + 4);
    d[0]=a.x; d[1]=a.y; d[2]=a.z; d[3]=a.w; d[4]=b.x; d[5]=b.y; d[6]=b.z; d[7]=b.w;
}
__device__ __forceinline__ void st8(float* p, const float* d) {
    *(float4*)p       = make_float4(d[0],d[1],d[2],d[3]);
    *(float4*)(p + 4) = make_float4(d[4],d[5],d[6],d[7]);
}

// Pack 8 fp32 weight mats into one bf16 [1024 x 512] (k = [IN | H]) and fuse biases.
__global__ void lstm_pack_weights(
    const float* Wif, const float* Wii, const float* Wio, const float* Wig,
    const float* Whf, const float* Whi, const float* Who, const float* Whg,
    const float* bif, const float* bii, const float* bio, const float* big,
    const float* bhf, const float* bhi, const float* bho, const float* bhg,
    unsigned short* W2, float* bias2)
{
    int id = blockIdx.x * 256 + threadIdx.x;   // 0..524287
    int n = id >> 9;
    int k = id & 511;
    int gate = n >> 8;
    int h = n & 255;
    const float* Wi = (gate==0)?Wif:(gate==1)?Wii:(gate==2)?Wio:Wig;
    const float* Wh = (gate==0)?Whf:(gate==1)?Whi:(gate==2)?Who:Whg;
    float v = (k < 256) ? Wi[h*256 + k] : Wh[h*256 + (k-256)];
    W2[id] = f2bf(v);
    if (id < 1024) {
        int g2 = id >> 8, h2 = id & 255;
        const float* bi = (g2==0)?bif:(g2==1)?bii:(g2==2)?bio:big;
        const float* bh = (g2==0)?bhf:(g2==1)?bhi:(g2==2)?bho:bhg;
        bias2[id] = bi[h2] + bh[h2];
    }
}

// TDM: DMA one K-chunk of B (1024 n-rows x 32 halves) from packed weights into LDS,
// padding 4 dwords after every 16 dwords -> LDS row stride 40 halves. ISA ch.8 D# layout.
__device__ __forceinline__ void tdm_load_b_chunk(const unsigned short* W2, unsigned lds_off, int kc) {
    unsigned long long ga = (unsigned long long)(uintptr_t)W2 + (unsigned)kc * 64u;
    u32x4 g0;
    g0[0] = 1u;                                                   // count=1, user mode
    g0[1] = lds_off;                                              // lds_addr (bytes)
    g0[2] = (unsigned)ga;                                         // global_addr[31:0]
    g0[3] = (unsigned)((ga >> 32) & 0x01FFFFFFu) | 0x80000000u;   // addr[56:32] | type=2
    u32x8 g1;
    g1[0] = (1u << 16)      // data_size = 2 bytes
          | (1u << 20)      // pad_enable
          | (3u << 22)      // pad_interval: 16 dwords (one 64B tile row)
          | (3u << 25);     // pad_amount: 4 dwords (8 halves)
    g1[1] = (512u << 16);   // tensor_dim0 = 512  (low 16 -> bits[63:48])
    g1[2] = (1024u << 16);  // tensor_dim0 hi = 0; tensor_dim1 = 1024 (low 16)
    g1[3] = (32u << 16);    // tensor_dim1 hi = 0; tile_dim0 = 32
    g1[4] = 1024u;          // tile_dim1 = 1024; tile_dim2 = 0
    g1[5] = 512u;           // tensor_dim0_stride[31:0] = 512 elements
    g1[6] = 0u;             // stride0 hi, stride1 lo
    g1[7] = 0u;
    asm volatile("tensor_load_to_lds %0, %1" :: "s"(g0), "s"(g1) : "memory");
}

// Stage A: 64 rows x 32 k fp32 -> bf16; 4 elements per thread (512 threads).
__device__ __forceinline__ void stage_A(const float* __restrict__ input,
                                        const float* __restrict__ hidden,
                                        int m0, int tid, int kc, unsigned short* lds_a) {
    int row = tid >> 3;
    int col = (tid & 7) * 4;
    int kg = kc * 32 + col;
    const float* src = (kg < 256) ? (input  + (size_t)(m0 + row) * 256 + kg)
                                  : (hidden + (size_t)(m0 + row) * 256 + (kg - 256));
    float4 v = *(const float4*)src;
    unsigned p0 = ((unsigned)f2bf(v.y) << 16) | f2bf(v.x);
    unsigned p1 = ((unsigned)f2bf(v.w) << 16) | f2bf(v.z);
    *(uint2*)(lds_a + row * LDS_A_STRIDE + col) = make_uint2(p0, p1);
}

// Fused: gates GEMM (bf16 WMMA, TDM-fed double-buffered LDS) + LSTM pointwise + log_softmax.
// 512 threads = 16 waves: 4 m-waves (16 rows) x 4 n-waves (256 cols) -> 16 acc tiles/wave.
__global__ void __launch_bounds__(THREADS, 1)
lstm_fused(const float* __restrict__ input, const float* __restrict__ hidden,
           const float* __restrict__ cellp, const unsigned short* __restrict__ W2,
           const float* __restrict__ bias2, float* __restrict__ out)
{
    extern __shared__ char smem[];
    float* gatesL = (float*)smem;   // 64 x 1032 f32, overlays staging after GEMM

    const int tid  = threadIdx.x;
    const int lane = tid & 31;
    const int wave = tid >> 5;
    const int wm   = wave & 3;    // 0..3 : 16-row slab
    const int wn   = wave >> 2;   // 0..3 : 256-col slab
    const int m0   = blockIdx.x * BROWS;
    const int half = lane >> 4;
    const int l15  = lane & 15;

    // Accumulators seeded with fused bias (constant down the rows dim).
    v8f acc[16];
#pragma unroll
    for (int nt = 0; nt < 16; ++nt) {
        float bv = bias2[wn * 256 + nt * 16 + l15];
        v8f c;
#pragma unroll
        for (int j = 0; j < 8; ++j) c[j] = bv;
        acc[nt] = c;
    }

    // Prologue: stage chunk 0 into buffer 0.
    stage_A(input, hidden, m0, tid, 0, (unsigned short*)smem);
    if (wave == 0)
        tdm_load_b_chunk(W2, (unsigned)(uintptr_t)(smem + B_OFF), 0);

    for (int kc = 0; kc < 16; ++kc) {
        const int cur = kc & 1;
        const int nxt = cur ^ 1;
        if (kc < 15) {
            // Stage chunk kc+1 into the other buffer; overlaps with this chunk's WMMAs.
            stage_A(input, hidden, m0, tid, kc + 1,
                    (unsigned short*)(smem + nxt * A_BYTES));
            if (wave == 0) {
                tdm_load_b_chunk(W2, (unsigned)(uintptr_t)(smem + B_OFF + nxt * B_BYTES), kc + 1);
                __builtin_amdgcn_s_wait_tensorcnt(1);   // chunk kc done; kc+1 may fly
            }
        } else {
            if (wave == 0)
                __builtin_amdgcn_s_wait_tensorcnt(0);
        }
        __syncthreads();   // chunk kc (A stores + TDM B) visible to all waves

        const unsigned short* la = (const unsigned short*)(smem + cur * A_BYTES);
        const unsigned short* lb = (const unsigned short*)(smem + B_OFF + cur * B_BYTES);

        // ---- A fragment (16x32 bf16)
        union V16 { uint4 q[2]; v16bf v; } af;
        {
            int row = wm * 16 + l15;
            const uint4* pa = (const uint4*)(la + row * LDS_A_STRIDE + half * 16);
            af.q[0] = pa[0];
            af.q[1] = pa[1];   // +16B
        }
        // ---- 16 WMMAs, B fragments 4-deep pipelined
        union V16 bb[4];
#pragma unroll
        for (int p = 0; p < 4; ++p) {
            int n = wn * 256 + p * 16 + l15;
            const uint4* pb = (const uint4*)(lb + n * LDS_B_STRIDE + half * 16);
            bb[p].q[0] = pb[0]; bb[p].q[1] = pb[1];
        }
#pragma unroll
        for (int nt = 0; nt < 16; ++nt) {
            acc[nt] = __builtin_amdgcn_wmma_f32_16x16x32_bf16(
                false, af.v, false, bb[nt & 3].v, (short)0, acc[nt], false, false);
            if (nt + 4 < 16) {
                int n = wn * 256 + (nt + 4) * 16 + l15;
                const uint4* pb = (const uint4*)(lb + n * LDS_B_STRIDE + half * 16);
                bb[nt & 3].q[0] = pb[0]; bb[nt & 3].q[1] = pb[1];
            }
        }
        __syncthreads();   // all reads of buffer `cur` done before it is overwritten
    }

    // ---- spill gates to LDS so f/i/o/g (different waves) combine per (row,h)
#pragma unroll
    for (int nt = 0; nt < 16; ++nt) {
        int n = wn * 256 + nt * 16 + l15;
#pragma unroll
        for (int r = 0; r < 8; ++r) {
            int m = wm * 16 + half * 8 + r;
            gatesL[m * GATE_STRIDE + n] = acc[nt][r];
        }
    }
    __syncthreads();

    // ---- pointwise LSTM + log_softmax: wave handles 4 rows, lane handles 8 h's
    const int mrow = wave * 4;
    const int h0   = lane * 8;
    for (int rr = 0; rr < 4; ++rr) {
        int m  = mrow + rr;
        int gr = m0 + m;
        const float* grow = gatesL + m * GATE_STRIDE;
        float fv[8], iv[8], ov[8], gv[8], cv[8], nh[8], nc[8];
        ld8(grow + h0,        fv);
        ld8(grow + 256 + h0,  iv);
        ld8(grow + 512 + h0,  ov);
        ld8(grow + 768 + h0,  gv);
        ld8(cellp + (size_t)gr * 256 + h0, cv);
        float mx = -3.0e38f;
#pragma unroll
        for (int j = 0; j < 8; ++j) {
            float ff = 1.0f / (1.0f + __expf(-fv[j]));
            float ii = 1.0f / (1.0f + __expf(-iv[j]));
            float oo = 1.0f / (1.0f + __expf(-ov[j]));
            float gg = tanhf(gv[j]);
            float c2 = ff * cv[j] + ii * gg;
            float h2 = oo * tanhf(c2);
            nc[j] = c2; nh[j] = h2;
            mx = fmaxf(mx, h2);
        }
#pragma unroll
        for (int off = 16; off > 0; off >>= 1) mx = fmaxf(mx, __shfl_xor(mx, off, 32));
        float s = 0.0f;
#pragma unroll
        for (int j = 0; j < 8; ++j) s += __expf(nh[j] - mx);
#pragma unroll
        for (int off = 16; off > 0; off >>= 1) s += __shfl_xor(s, off, 32);
        float lz = mx + __logf(s);
        float lsm[8];
#pragma unroll
        for (int j = 0; j < 8; ++j) lsm[j] = nh[j] - lz;

        size_t base = (size_t)gr * 256 + h0;
        st8(out + base,          lsm);   // log_softmax
        st8(out + BH + base,     nh);    // new_hidden
        st8(out + 2 * BH + base, nc);    // new_cell
    }
}

extern "C" void kernel_launch(void* const* d_in, const int* in_sizes, int n_in,
                              void* d_out, int out_size, void* d_ws, size_t ws_size,
                              hipStream_t stream) {
    const float* input  = (const float*)d_in[0];
    const float* hidden = (const float*)d_in[1];
    const float* cell   = (const float*)d_in[2];
    const float* Wif = (const float*)d_in[3];
    const float* bif = (const float*)d_in[4];
    const float* Wii = (const float*)d_in[5];
    const float* bii = (const float*)d_in[6];
    const float* Wio = (const float*)d_in[7];
    const float* bio = (const float*)d_in[8];
    const float* Wig = (const float*)d_in[9];
    const float* big = (const float*)d_in[10];
    const float* Whf = (const float*)d_in[11];
    const float* bhf = (const float*)d_in[12];
    const float* Whi = (const float*)d_in[13];
    const float* bhi = (const float*)d_in[14];
    const float* Who = (const float*)d_in[15];
    const float* bho = (const float*)d_in[16];
    const float* Whg = (const float*)d_in[17];
    const float* bhg = (const float*)d_in[18];

    unsigned short* W2 = (unsigned short*)d_ws;                       // 1 MB bf16
    float* bias2 = (float*)((char*)d_ws + (size_t)NG * K_DIM * 2);    // 4 KB

    lstm_pack_weights<<<2048, 256, 0, stream>>>(
        Wif, Wii, Wio, Wig, Whf, Whi, Who, Whg,
        bif, bii, bio, big, bhf, bhi, bho, bhg, W2, bias2);

    lstm_fused<<<65536 / BROWS, THREADS, SMEM_BYTES, stream>>>(
        input, hidden, cell, W2, bias2, (float*)d_out);
}
